// NeRF_MVS_78477642432658
// MI455X (gfx1250) — compile-verified, matrix operand order checked by reading, not run on hardware
//
#include <hip/hip_runtime.h>
#include <hip/hip_bf16.h>

// ---------------------------------------------------------------------------
// NeRF MLP fused for MI455X (gfx1250): wave32 WMMA f16 pipeline, v2.
//  - Each wave owns 32 rows (two M=16 tiles) -> every B fragment fetched from
//    L2 feeds 2 WMMAs (halves the dominant L2 weight traffic).
//  - A fragments for the whole layer are hoisted into VGPRs (16x fewer LDS
//    reads than reloading per N-tile).
//  - Everything is wave-local: zero barriers.
// ---------------------------------------------------------------------------

typedef __attribute__((ext_vector_type(16))) _Float16 v16h;
typedef __attribute__((ext_vector_type(8)))  float    v8f;

#define BIGK (1 << 30)

// Packed-weight offsets in halves (fragment = 32 lanes * 16 halves = 512)
static constexpr size_t OFF_W0 = 0;                       //  2*16 frags
static constexpr size_t OFF_W1 = OFF_W0 + 2  * 16 * 512;  //  8*16
static constexpr size_t OFF_W2 = OFF_W1 + 8  * 16 * 512;
static constexpr size_t OFF_W3 = OFF_W2 + 8  * 16 * 512;
static constexpr size_t OFF_W4 = OFF_W3 + 8  * 16 * 512;  // 10*16 (row-permuted)
static constexpr size_t OFF_W5 = OFF_W4 + 10 * 16 * 512;
static constexpr size_t OFF_W6 = OFF_W5 + 8  * 16 * 512;
static constexpr size_t OFF_W7 = OFF_W6 + 8  * 16 * 512;
static constexpr size_t OFF_WF = OFF_W7 + 8  * 16 * 512;
static constexpr size_t OFF_WD = OFF_WF + 8  * 16 * 512;  //  9*8

// LDS geometry (halves). 256 rows per block.
static constexpr int ROWS    = 256;
static constexpr int LDS_IN  = 0;                       // 256 x 64
static constexpr int LDS_DIR = LDS_IN  + ROWS * 64;     // 256 x 32
static constexpr int LDS_H0  = LDS_DIR + ROWS * 32;     // 256 x 256
static constexpr int LDS_H1  = LDS_H0  + ROWS * 256;    // 256 x 256
static constexpr int LDS_TOT = LDS_H1  + ROWS * 256;    // 155648 halves = 304 KB

// ---------------------------------------------------------------------------
// Weight repack: fp32 row-major W[fin][fout] -> f16 fragments.
// Fragment (nt,kt): lane L holds column n = nt*16 + (L&15),
//                   halves e=0..15 hold k = kt*32 + (L>>4)*16 + e.
// Packed K-row kp maps to source row:  kp <  mainLen -> kp + headOff
//                                      kp >= mainLen -> kp - mainLen (< headLen)
// (rotates the layer-4 skip concat so the 63-wide xyz lands K-aligned)
// ---------------------------------------------------------------------------
__global__ void cvt_weight(const float* __restrict__ src, _Float16* __restrict__ dst,
                           int fin, int fout, int KT, int NT,
                           int mainLen, int headOff, int headLen, int total)
{
  for (int i = blockIdx.x * blockDim.x + threadIdx.x; i < total;
       i += gridDim.x * blockDim.x) {
    int e    = i & 15;
    int lane = (i >> 4) & 31;
    int frag = i >> 9;
    int kt   = frag % KT;
    int nt   = frag / KT;
    int n    = nt * 16 + (lane & 15);
    int kp   = kt * 32 + ((lane >> 4) << 4) + e;
    int r; bool ok;
    if (kp < mainLen) { r = kp + headOff; ok = (r < fin); }
    else              { r = kp - mainLen; ok = (r < headLen); }
    ok = ok && (n < fout);
    float v = ok ? src[(size_t)r * fout + n] : 0.0f;
    dst[i] = (_Float16)v;
  }
}

// ---------------------------------------------------------------------------
// Fragment loaders
// ---------------------------------------------------------------------------
__device__ __forceinline__ v16h loadA(const _Float16* base, int rowBase, int ld,
                                      int k0, int lane)
{
  const _Float16* p = base + (size_t)(rowBase + (lane & 15)) * ld
                           + k0 + ((lane >> 4) << 4);
  return *(const v16h*)p;
}

__device__ __forceinline__ v16h loadB(const _Float16* w, int frag, int lane)
{
  return *(const v16h*)(w + ((size_t)frag * 32 + lane) * 16);  // 32B/lane contiguous
}

__device__ __forceinline__ v8f wmma_f16(v16h a, v16h b, v8f c)
{
  return __builtin_amdgcn_wmma_f32_16x16x32_f16(false, a, false, b,
                                                (short)0, c, false, false);
}

// ---------------------------------------------------------------------------
// Dense layer, MT=2 row-tiles per wave, A hoisted in VGPRs for whole layer.
//   out[32 x NT*16] = act( in[32 x KT*32] @ W + b )
// ---------------------------------------------------------------------------
template<int KT, bool RELU>
__device__ void layer_dense2(const _Float16* inL, int ldIn,
                             const _Float16* __restrict__ W,
                             const float* __restrict__ bias,
                             _Float16* outL, int ldOut, int NT,
                             int rowBase, int lane)
{
  v16h a[2][KT];
#pragma unroll
  for (int mt = 0; mt < 2; ++mt)
#pragma unroll
    for (int kt = 0; kt < KT; ++kt)
      a[mt][kt] = loadA(inL, rowBase + mt * 16, ldIn, kt * 32, lane);

  for (int nt = 0; nt < NT; ++nt) {
    if (nt + 1 < NT)
      __builtin_prefetch(W + ((size_t)(nt + 1) * KT * 32 + lane) * 16, 0, 0);
    v8f acc0 = {}, acc1 = {};
#pragma unroll
    for (int kt = 0; kt < KT; ++kt) {
      v16h b = loadB(W, nt * KT + kt, lane);
      acc0 = wmma_f16(a[0][kt], b, acc0);
      acc1 = wmma_f16(a[1][kt], b, acc1);
    }
    int   col = nt * 16 + (lane & 15);
    float bv  = bias[col];
#pragma unroll
    for (int r = 0; r < 8; ++r) {
      int   m  = r + ((lane >> 4) << 3);           // C/D layout: M = r + 8*(lane>=16)
      float v0 = acc0[r] + bv;
      float v1 = acc1[r] + bv;
      if (RELU) { v0 = fmaxf(v0, 0.0f); v1 = fmaxf(v1, 0.0f); }
      outL[(size_t)(rowBase + m)      * ldOut + col] = (_Float16)v0;
      outL[(size_t)(rowBase + 16 + m) * ldOut + col] = (_Float16)v1;
    }
  }
}

// Concat layer: first KT1 K-tiles from region1 (A hoisted), last KT2 from
// region2 (small; streamed from LDS inside the N-loop).
template<int KT1, int KT2, bool RELU>
__device__ void layer_concat2(const _Float16* in1, int ld1,
                              const _Float16* in2, int ld2,
                              const _Float16* __restrict__ W,
                              const float* __restrict__ bias,
                              _Float16* outL, int ldOut, int NT,
                              int rowBase, int lane)
{
  constexpr int KT = KT1 + KT2;
  v16h a[2][KT1];
#pragma unroll
  for (int mt = 0; mt < 2; ++mt)
#pragma unroll
    for (int kt = 0; kt < KT1; ++kt)
      a[mt][kt] = loadA(in1, rowBase + mt * 16, ld1, kt * 32, lane);

  for (int nt = 0; nt < NT; ++nt) {
    if (nt + 1 < NT)
      __builtin_prefetch(W + ((size_t)(nt + 1) * KT * 32 + lane) * 16, 0, 0);
    v8f acc0 = {}, acc1 = {};
#pragma unroll
    for (int kt = 0; kt < KT1; ++kt) {
      v16h b = loadB(W, nt * KT + kt, lane);
      acc0 = wmma_f16(a[0][kt], b, acc0);
      acc1 = wmma_f16(a[1][kt], b, acc1);
    }
#pragma unroll
    for (int kt = 0; kt < KT2; ++kt) {
      v16h b  = loadB(W, nt * KT + KT1 + kt, lane);
      v16h a0 = loadA(in2, rowBase,      ld2, kt * 32, lane);
      v16h a1 = loadA(in2, rowBase + 16, ld2, kt * 32, lane);
      acc0 = wmma_f16(a0, b, acc0);
      acc1 = wmma_f16(a1, b, acc1);
    }
    int   col = nt * 16 + (lane & 15);
    float bv  = bias[col];
#pragma unroll
    for (int r = 0; r < 8; ++r) {
      int   m  = r + ((lane >> 4) << 3);
      float v0 = acc0[r] + bv;
      float v1 = acc1[r] + bv;
      if (RELU) { v0 = fmaxf(v0, 0.0f); v1 = fmaxf(v1, 0.0f); }
      outL[(size_t)(rowBase + m)      * ldOut + col] = (_Float16)v0;
      outL[(size_t)(rowBase + 16 + m) * ldOut + col] = (_Float16)v1;
    }
  }
}

// ---------------------------------------------------------------------------
// Fused NeRF kernel: 8 waves * 32 rows = 256 rows per block; 304 KB LDS.
// ---------------------------------------------------------------------------
__global__ __launch_bounds__(256)
void nerf_fused(const float* __restrict__ x,
                const _Float16* __restrict__ wp,
                const float* __restrict__ b0, const float* __restrict__ b1,
                const float* __restrict__ b2, const float* __restrict__ b3,
                const float* __restrict__ b4, const float* __restrict__ b5,
                const float* __restrict__ b6, const float* __restrict__ b7,
                const float* __restrict__ bf, const float* __restrict__ bd,
                const float* __restrict__ wsv, const float* __restrict__ bs,
                const float* __restrict__ wrv, const float* __restrict__ br,
                float* __restrict__ out)
{
  extern __shared__ _Float16 smem[];
  _Float16* lds_in  = smem + LDS_IN;    // 256 x 64  (xyz, col 63 = pad)
  _Float16* lds_dir = smem + LDS_DIR;   // 256 x 32  (dir, cols 30,31 = pad)
  _Float16* lds_h0  = smem + LDS_H0;    // 256 x 256
  _Float16* lds_h1  = smem + LDS_H1;    // 256 x 256

  const int  lane    = threadIdx.x & 31;
  const int  wave    = threadIdx.x >> 5;
  const int  rowBase = wave * 32;
  const long gRow0   = (long)blockIdx.x * ROWS;

  // ---- stage this wave's 32 input rows into LDS (f32 -> f16) ----
  for (int r = 0; r < 32; ++r) {
    int row = rowBase + r;
    const float* xr = x + (gRow0 + row) * 93;          // 63 xyz + 30 dir
    for (int c = lane; c < 95; c += 32) {
      float v = (c < 93) ? xr[c] : 0.0f;
      if (c < 63) lds_in [row * 64 + c]        = (_Float16)v;
      else        lds_dir[row * 32 + (c - 63)] = (_Float16)v;
    }
    if (lane == 0) lds_in[row * 64 + 63] = (_Float16)0.0f;
  }

  // ---- trunk: 8 layers, skip-concat at layer 4 (wave-local, no barriers) ----
  layer_dense2<2, true>(lds_in, 64,  wp + OFF_W0, b0, lds_h0, 256, 16, rowBase, lane);
  layer_dense2<8, true>(lds_h0, 256, wp + OFF_W1, b1, lds_h1, 256, 16, rowBase, lane);
  layer_dense2<8, true>(lds_h1, 256, wp + OFF_W2, b2, lds_h0, 256, 16, rowBase, lane);
  layer_dense2<8, true>(lds_h0, 256, wp + OFF_W3, b3, lds_h1, 256, 16, rowBase, lane);
  // w4 was row-rotated at repack: K-tiles 0..7 = h, K-tiles 8..9 = xyz(+pad)
  layer_concat2<8, 2, true>(lds_h1, 256, lds_in, 64,
                            wp + OFF_W4, b4, lds_h0, 256, 16, rowBase, lane);
  layer_dense2<8, true>(lds_h0, 256, wp + OFF_W5, b5, lds_h1, 256, 16, rowBase, lane);
  layer_dense2<8, true>(lds_h1, 256, wp + OFF_W6, b6, lds_h0, 256, 16, rowBase, lane);
  layer_dense2<8, true>(lds_h0, 256, wp + OFF_W7, b7, lds_h1, 256, 16, rowBase, lane);

  // ---- sigma head: h(256) -> 1; every lane owns one of the wave's 32 rows.
  // k is staggered per lane so concurrent lanes hit different LDS banks.
  {
    int row = rowBase + lane;
    float s = bs[0];
    int k0 = (lane * 8) & 255;
    for (int i = 0; i < 256; ++i) {
      int k = (k0 + i) & 255;
      s += (float)lds_h1[row * 256 + k] * wsv[k];
    }
    out[(gRow0 + row) * 4 + 3] = s;
  }

  // ---- xyz_final = h @ wf + bf (no relu) ----
  layer_dense2<8, false>(lds_h1, 256, wp + OFF_WF, bf, lds_h0, 256, 16, rowBase, lane);

  // ---- d = relu([xyz_final, dir] @ wd + bd): K = 256 + 32(pad), N = 128 ----
  // output stride kept at 256 so each wave's region stays disjoint.
  layer_concat2<8, 1, true>(lds_h0, 256, lds_dir, 32,
                            wp + OFF_WD, bd, lds_h1, 256, 8, rowBase, lane);

  // ---- rgb head: d(128) -> 3 with sigmoid; one row per lane, staggered k ----
  {
    int row = rowBase + lane;
    float r0 = br[0], r1 = br[1], r2 = br[2];
    int k0 = (lane * 4) & 127;
    for (int i = 0; i < 128; ++i) {
      int k = (k0 + i) & 127;
      float dv = (float)lds_h1[row * 256 + k];
      r0 += dv * wrv[k * 3 + 0];
      r1 += dv * wrv[k * 3 + 1];
      r2 += dv * wrv[k * 3 + 2];
    }
    float* o = out + (gRow0 + row) * 4;
    o[0] = 1.0f / (1.0f + __expf(-r0));
    o[1] = 1.0f / (1.0f + __expf(-r1));
    o[2] = 1.0f / (1.0f + __expf(-r2));
  }
}

// ---------------------------------------------------------------------------
extern "C" void kernel_launch(void* const* d_in, const int* in_sizes, int n_in,
                              void* d_out, int out_size, void* d_ws, size_t ws_size,
                              hipStream_t stream)
{
  const float* x = (const float*)d_in[0];
  const float* w[8]; const float* b[8];
  for (int i = 0; i < 8; ++i) {
    w[i] = (const float*)d_in[1 + 2 * i];
    b[i] = (const float*)d_in[2 + 2 * i];
  }
  const float* wf  = (const float*)d_in[17];
  const float* bf  = (const float*)d_in[18];
  const float* wd  = (const float*)d_in[19];
  const float* bd  = (const float*)d_in[20];
  const float* wsv = (const float*)d_in[21];
  const float* bs  = (const float*)d_in[22];
  const float* wr  = (const float*)d_in[23];
  const float* br  = (const float*)d_in[24];

  _Float16* wp = (_Float16*)d_ws;   // ~1.19 MB of packed f16 weights

  auto cvt = [&](const float* s, size_t off, int fin, int fout, int KT, int NT,
                 int mL, int hO, int hL) {
    int total = NT * KT * 512;
    hipLaunchKernelGGL(cvt_weight, dim3((total + 255) / 256), dim3(256), 0, stream,
                       s, wp + off, fin, fout, KT, NT, mL, hO, hL, total);
  };

  cvt(w[0], OFF_W0,  63, 256,  2, 16, BIGK, 0, 0);
  cvt(w[1], OFF_W1, 256, 256,  8, 16, BIGK, 0, 0);
  cvt(w[2], OFF_W2, 256, 256,  8, 16, BIGK, 0, 0);
  cvt(w[3], OFF_W3, 256, 256,  8, 16, BIGK, 0, 0);
  // w4 (319 x 256): rotate rows so packed K = [h(256) | xyz(63) | pad]
  cvt(w[4], OFF_W4, 319, 256, 10, 16, 256, 63, 63);
  cvt(w[5], OFF_W5, 256, 256,  8, 16, BIGK, 0, 0);
  cvt(w[6], OFF_W6, 256, 256,  8, 16, BIGK, 0, 0);
  cvt(w[7], OFF_W7, 256, 256,  8, 16, BIGK, 0, 0);
  cvt(wf,   OFF_WF, 256, 256,  8, 16, BIGK, 0, 0);
  cvt(wd,   OFF_WD, 286, 128,  9,  8, BIGK, 0, 0);  // rows 256..285 = dir part

  const size_t shmem = (size_t)LDS_TOT * sizeof(_Float16);   // 304 KB
  (void)hipFuncSetAttribute((const void*)nerf_fused,
                            hipFuncAttributeMaxDynamicSharedMemorySize,
                            (int)shmem);
  hipLaunchKernelGGL(nerf_fused, dim3(262144 / ROWS), dim3(256), shmem, stream,
                     x, wp,
                     b[0], b[1], b[2], b[3], b[4], b[5], b[6], b[7],
                     bf, bd, wsv, bs, wr, br, (float*)d_out);
}
